// SelfAttention_16484084482476
// MI455X (gfx1250) — matrix operand order repaired and checked by reading, not compiled, not run on hardware
//
#include <hip/hip_runtime.h>

typedef _Float16 half_t;
typedef __attribute__((ext_vector_type(16))) _Float16 v16h;
typedef __attribute__((ext_vector_type(8)))  _Float16 v8h;   // 16 bytes
typedef __attribute__((ext_vector_type(4)))  _Float16 v4h;
typedef __attribute__((ext_vector_type(8)))  float    v8f;
typedef __attribute__((ext_vector_type(4)))  int      v4i;

typedef __attribute__((address_space(1))) v4i gv4i;  // global int4
typedef __attribute__((address_space(3))) v4i lv4i;  // LDS int4

#define BATCH    8
#define CCH      256
#define CQ       32
#define NSP      4096
#define QKV_ROWS 320   // 32 q + 32 k + 256 v

// ---------------- 16-byte global -> LDS copy (async if available) ------------

#if defined(__has_builtin)
#if __has_builtin(__builtin_amdgcn_global_load_async_to_lds_b128)
#define HAVE_ASYNC_LDS 1
#endif
#if __has_builtin(__builtin_amdgcn_s_wait_asynccnt)
#define HAVE_WAIT_ASYNC_BUILTIN 1
#endif
#endif

__device__ __forceinline__ void cp16(half_t* dst, const half_t* src) {
#ifdef HAVE_ASYNC_LDS
  __builtin_amdgcn_global_load_async_to_lds_b128(
      (gv4i*)(src), (lv4i*)(dst), 0, 0);
#else
  *reinterpret_cast<v8h*>(dst) = *reinterpret_cast<const v8h*>(src);
#endif
}

__device__ __forceinline__ void wait_cp16() {
#ifdef HAVE_ASYNC_LDS
#ifdef HAVE_WAIT_ASYNC_BUILTIN
  __builtin_amdgcn_s_wait_asynccnt(0);
#else
  asm volatile("s_wait_asynccnt 0" ::: "memory");
#endif
#endif
}

// ---------------- WMMA fragment loaders (CDNA5 16-bit layouts, ISA 7.12.2) ----

// A-matrix 16x32 f16, A[m][k] at base[m*ld + k].
__device__ __forceinline__ v16h load_A16x32(const half_t* base, int ld, int lane) {
  const int h = lane >> 4;
  const int m = lane & 15;
  v16h a;
#pragma unroll
  for (int g = 0; g < 8; ++g) {
    int k = ((g < 4) ? (2 * g) : (16 + 2 * (g - 4))) + 8 * h;
    a[2 * g]     = base[m * ld + k];
    a[2 * g + 1] = base[m * ld + k + 1];
  }
  return a;
}

// B-matrix 32x16 f16, B[k][n] at base[k*ld + n]. lane%16 = N column; elements = K (same order as A).
__device__ __forceinline__ v16h load_B32x16(const half_t* base, int ld, int lane) {
  const int h = lane >> 4;
  const int n = lane & 15;
  v16h b;
#pragma unroll
  for (int g = 0; g < 8; ++g) {
    int k = ((g < 4) ? (2 * g) : (16 + 2 * (g - 4))) + 8 * h;
    b[2 * g]     = base[k * ld + n];
    b[2 * g + 1] = base[(k + 1) * ld + n];
  }
  return b;
}

// ---------------- prep kernels -----------------------------------------------

__global__ void prep_weights(const float* __restrict__ Wq, const float* __restrict__ Wk,
                             const float* __restrict__ Wv, const float* __restrict__ bq,
                             const float* __restrict__ bk, const float* __restrict__ bv,
                             half_t* __restrict__ Wh, float* __restrict__ bias) {
  int i = blockIdx.x * blockDim.x + threadIdx.x;
  if (i < QKV_ROWS * CCH) {
    int o = i / CCH, c = i - o * CCH;
    float w;
    if (o < 32)       w = Wq[o * CCH + c];
    else if (o < 64)  w = Wk[(o - 32) * CCH + c];
    else              w = Wv[(o - 64) * CCH + c];
    Wh[i] = (half_t)w;
  }
  if (i < QKV_ROWS) {
    bias[i] = (i < 32) ? bq[i] : (i < 64) ? bk[i - 32] : bv[i - 64];
  }
}

__global__ void conv_x4(const float* __restrict__ x, half_t* __restrict__ xh, int n4) {
  int i = blockIdx.x * blockDim.x + threadIdx.x;
  if (i < n4) {
    float4 v = reinterpret_cast<const float4*>(x)[i];
    v4h h;
    h[0] = (half_t)v.x; h[1] = (half_t)v.y; h[2] = (half_t)v.z; h[3] = (half_t)v.w;
    reinterpret_cast<v4h*>(xh)[i] = h;
  }
}

// ---------------- fused QKV projection GEMM (double-buffered async) ----------
// qkv[b][o][n] = sum_c Wh[o][c] * xh[b][c][n] + bias[o]   (o in [0,320))

__global__ __launch_bounds__(256) void qkv_gemm(const half_t* __restrict__ Wh,
                                                const float*  __restrict__ bias,
                                                const half_t* __restrict__ xh,
                                                half_t* __restrict__ qkv) {
  __shared__ __align__(16) half_t As[2][64][40];  // [o][k] (row = 80B)
  __shared__ __align__(16) half_t Bs[2][32][72];  // [k][n] (row = 144B)

  const int tid  = threadIdx.x;
  const int lane = tid & 31;
  const int wave = tid >> 5;
  const int n0 = blockIdx.x * 64;
  const int o0 = blockIdx.y * 64;
  const int b  = blockIdx.z;

  const int t0 = wave * 2, t1 = wave * 2 + 1;
  const int ot0 = t0 >> 2, nt0 = t0 & 3;
  const int ot1 = t1 >> 2, nt1 = t1 & 3;

  // per-thread staging coordinates (one 16B segment each)
  const int ar = tid >> 2, aseg = (tid & 3) * 8;   // A: 64 rows x 4 segs
  const int br = tid >> 3, bseg = (tid & 7) * 8;   // B: 32 rows x 8 segs

  // prologue: stage k-block 0 into buffer 0
  cp16(&As[0][ar][aseg], &Wh[(o0 + ar) * CCH + aseg]);
  cp16(&Bs[0][br][bseg], &xh[((size_t)b * CCH + br) * NSP + n0 + bseg]);

  v8f acc0 = {}, acc1 = {};

  for (int i = 0; i < CCH / 32; ++i) {
    const int cur = i & 1;
    wait_cp16();          // current buffer's async copies complete
    __syncthreads();      // visible to all waves; prior reads of other buffer done

    if (i + 1 < CCH / 32) {  // stream next k-block while computing
      const int kk = 32 * (i + 1);
      cp16(&As[cur ^ 1][ar][aseg], &Wh[(o0 + ar) * CCH + kk + aseg]);
      cp16(&Bs[cur ^ 1][br][bseg], &xh[((size_t)b * CCH + kk + br) * NSP + n0 + bseg]);
    }

    v16h a  = load_A16x32(&As[cur][ot0 * 16][0], 40, lane);
    v16h bm = load_B32x16(&Bs[cur][0][nt0 * 16], 72, lane);
    acc0 = __builtin_amdgcn_wmma_f32_16x16x32_f16(false, a, false, bm, (short)0, acc0, false, false);
    a  = load_A16x32(&As[cur][ot1 * 16][0], 40, lane);
    bm = load_B32x16(&Bs[cur][0][nt1 * 16], 72, lane);
    acc1 = __builtin_amdgcn_wmma_f32_16x16x32_f16(false, a, false, bm, (short)0, acc1, false, false);
  }

  const int h = lane >> 4, nn = lane & 15;
#pragma unroll
  for (int g = 0; g < 8; ++g) {
    int o = o0 + ot0 * 16 + g + 8 * h;
    int n = n0 + nt0 * 16 + nn;
    qkv[((size_t)b * QKV_ROWS + o) * NSP + n] = (half_t)(acc0[g] + bias[o]);
    o = o0 + ot1 * 16 + g + 8 * h;
    n = n0 + nt1 * 16 + nn;
    qkv[((size_t)b * QKV_ROWS + o) * NSP + n] = (half_t)(acc1[g] + bias[o]);
  }
}

// ---------------- fused flash attention (double-buffered V stream) -----------
// out[c,m] = gamma * (sum_n v[c,n] softmax_n(q.n dot k.m)) / L[m] + x[c,m]

__global__ __launch_bounds__(256) void attn_fused(const half_t* __restrict__ qkv,
                                                  const float*  __restrict__ x,
                                                  const float*  __restrict__ gamma,
                                                  float* __restrict__ out) {
  __shared__ __align__(16) half_t kT[CQ][72];      // [c][m], persistent
  __shared__ __align__(16) half_t qT[64][40];      // [n][c]
  __shared__ __align__(16) half_t Vs[2][CCH][72];  // [c][n], double-buffered
  __shared__ __align__(16) float  Ss[64][65];      // scores
  __shared__ __align__(16) half_t Ps[64][72];      // exp probs
  __shared__ float runmax[64], runL[64], rscale[64];

  const int tid  = threadIdx.x;
  const int lane = tid & 31;
  const int wave = tid >> 5;
  const int m0 = blockIdx.x * 64;
  const int b  = blockIdx.y;
  const size_t qb = (size_t)b * QKV_ROWS * NSP;
  const int c0 = wave * 32;

  const int r8  = tid >> 3;            // 0..31
  const int s8  = (tid & 7) * 8;       // 16B segment within a 64-wide row

  if (tid < 64) { runmax[tid] = -1e30f; runL[tid] = 0.0f; }

  // prologue: persistent k tile + first v tile into buffer 0
  cp16(&kT[r8][s8], &qkv[qb + (size_t)(CQ + r8) * NSP + m0 + s8]);
#pragma unroll
  for (int it = 0; it < 8; ++it) {
    int c = it * 32 + r8;
    cp16(&Vs[0][c][s8], &qkv[qb + (size_t)(64 + c) * NSP + s8]);
  }

  v8f acc[2][4];
#pragma unroll
  for (int i = 0; i < 2; ++i)
#pragma unroll
    for (int j = 0; j < 4; ++j) acc[i][j] = (v8f){};

  for (int i = 0; i < NSP / 64; ++i) {
    const int n0  = i * 64;
    const int cur = i & 1;

    wait_cp16();  // current V tile (and kT on first iter) in LDS

    // stage q (transposed via registers, coalesced 16B global reads)
    {
      v8h q8 = *reinterpret_cast<const v8h*>(&qkv[qb + (size_t)r8 * NSP + n0 + s8]);
#pragma unroll
      for (int j = 0; j < 8; ++j) qT[s8 + j][r8] = q8[j];
      if (n0 + 64 < NSP)  // prefetch next q segment
        __builtin_prefetch(&qkv[qb + (size_t)r8 * NSP + n0 + 64 + s8], 0, 1);
    }
    __syncthreads();

    // stream next V tile into the other buffer while we compute on this one
    if (i + 1 < NSP / 64) {
#pragma unroll
      for (int it = 0; it < 8; ++it) {
        int c = it * 32 + r8;
        cp16(&Vs[cur ^ 1][c][s8], &qkv[qb + (size_t)(64 + c) * NSP + n0 + 64 + s8]);
      }
    }

    // S = q^T k  (16 tiles of 16x16 over (n,m); 2 per wave; K=32=Cq)
#pragma unroll
    for (int t = 0; t < 2; ++t) {
      int tile = wave * 2 + t;
      int nt = tile >> 2, mt = tile & 3;
      v16h a  = load_A16x32(&qT[nt * 16][0], 40, lane);
      v16h bm = load_B32x16(&kT[0][mt * 16], 72, lane);
      v8f s = {};
      s = __builtin_amdgcn_wmma_f32_16x16x32_f16(false, a, false, bm, (short)0, s, false, false);
      int h = lane >> 4, nn = lane & 15;
#pragma unroll
      for (int g = 0; g < 8; ++g)
        Ss[nt * 16 + g + 8 * h][mt * 16 + nn] = s[g];
    }
    __syncthreads();

    // online softmax over n (one thread per m column)
    if (tid < 64) {
      const int m = tid;
      float bmax = -1e30f;
#pragma unroll 8
      for (int n = 0; n < 64; ++n) bmax = fmaxf(bmax, Ss[n][m]);
      float om = runmax[m];
      float nm = fmaxf(om, bmax);
      float sc = __expf(om - nm);
      float l  = runL[m] * sc;
#pragma unroll 8
      for (int n = 0; n < 64; ++n) {
        float p = __expf(Ss[n][m] - nm);
        l += p;
        Ps[n][m] = (half_t)p;
      }
      runmax[m] = nm; runL[m] = l; rscale[m] = sc;
    }
    __syncthreads();

    // rescale accumulators and accumulate O += V * P
    v16h afr[2][2];
#pragma unroll
    for (int cs = 0; cs < 2; ++cs)
#pragma unroll
      for (int kc = 0; kc < 2; ++kc)
        afr[cs][kc] = load_A16x32(&Vs[cur][c0 + cs * 16][kc * 32], 72, lane);

#pragma unroll
    for (int mt = 0; mt < 4; ++mt) {
      float sc = rscale[mt * 16 + (lane & 15)];  // constant per lane across a frag
#pragma unroll
      for (int cs = 0; cs < 2; ++cs)
#pragma unroll
        for (int g = 0; g < 8; ++g) acc[cs][mt][g] *= sc;
#pragma unroll
      for (int kc = 0; kc < 2; ++kc) {
        v16h bm = load_B32x16(&Ps[kc * 32][mt * 16], 72, lane);
#pragma unroll
        for (int cs = 0; cs < 2; ++cs)
          acc[cs][mt] = __builtin_amdgcn_wmma_f32_16x16x32_f16(
              false, afr[cs][kc], false, bm, (short)0, acc[cs][mt], false, false);
      }
    }
    __syncthreads();
  }

  // epilogue: normalize, gamma-scale, residual add
  const float g0 = gamma[0];
  const int h = lane >> 4, nn = lane & 15;
#pragma unroll
  for (int cs = 0; cs < 2; ++cs)
#pragma unroll
    for (int mt = 0; mt < 4; ++mt) {
      float invl = 1.0f / runL[mt * 16 + nn];
#pragma unroll
      for (int g = 0; g < 8; ++g) {
        int c = c0 + cs * 16 + g + 8 * h;
        int m = mt * 16 + nn;
        size_t idx = ((size_t)b * CCH + c) * NSP + m0 + m;
        out[idx] = g0 * acc[cs][mt][g] * invl + x[idx];
      }
    }
}

// ---------------- launch -----------------------------------------------------

extern "C" void kernel_launch(void* const* d_in, const int* in_sizes, int n_in,
                              void* d_out, int out_size, void* d_ws, size_t ws_size,
                              hipStream_t stream) {
  const float* x     = (const float*)d_in[0];
  const float* Wq    = (const float*)d_in[1];
  const float* bq    = (const float*)d_in[2];
  const float* Wk    = (const float*)d_in[3];
  const float* bk    = (const float*)d_in[4];
  const float* Wv    = (const float*)d_in[5];
  const float* bv    = (const float*)d_in[6];
  const float* gamma = (const float*)d_in[7];
  float* out = (float*)d_out;

  char* ws = (char*)d_ws;
  half_t* Wh   = (half_t*)(ws);                       // 320*256*2 = 160 KB
  float*  bias = (float*)(ws + 164096);               // 320*4
  half_t* xh   = (half_t*)(ws + 262144);              // 16 MB
  half_t* qkv  = (half_t*)(ws + 262144 + (size_t)16 * 1024 * 1024); // 20 MB

  const int xtot = BATCH * CCH * NSP;

  prep_weights<<<(QKV_ROWS * CCH + 255) / 256, 256, 0, stream>>>(Wq, Wk, Wv, bq, bk, bv, Wh, bias);
  conv_x4<<<(xtot / 4 + 255) / 256, 256, 0, stream>>>(x, xh, xtot / 4);
  qkv_gemm<<<dim3(NSP / 64, QKV_ROWS / 64, BATCH), 256, 0, stream>>>(Wh, bias, xh, qkv);
  attn_fused<<<dim3(NSP / 64, BATCH), 256, 0, stream>>>(qkv, x, gamma, out);
}
